// VectorQuantizer_45621142618683
// MI455X (gfx1250) — matrix operand order repaired and checked by reading, not compiled, not run on hardware
//
#include <hip/hip_runtime.h>

#define D        256
#define NROWS    16384      // 16*32*32
#define KCODES   8192
#define KSPLIT   4          // K-dim split for occupancy; merged via atomic max on packed keys
#define VQ_EPS   1e-12f
#define VQ_BETA  0.25f

// Padded LDS row: 256 halves of code data + 8 halves pad -> 264 halves (528 B).
// Lane stride 132 dwords == 4 banks -> 16 lanes/half cover all 64 banks, no conflicts.
#define LROW     264

typedef __attribute__((ext_vector_type(16))) _Float16 v16h;
typedef __attribute__((ext_vector_type(8)))  _Float16 v8h;
typedef __attribute__((ext_vector_type(8)))  float    v8f;
typedef int v4i __attribute__((vector_size(4 * sizeof(int))));   // b128 payload type

typedef __attribute__((address_space(1))) v4i* gptr_v4i;         // global b128 ptr
typedef __attribute__((address_space(3))) v4i* lptr_v4i;         // LDS b128 ptr

union FragU { v16h v; v8h h[2]; };

#if __has_builtin(__builtin_amdgcn_global_load_async_to_lds_b128)
#define USE_ASYNC_LDS 1
#else
#define USE_ASYNC_LDS 0
#endif

#if __has_builtin(__builtin_amdgcn_s_wait_asynccnt)
#define WAIT_ASYNC(n) __builtin_amdgcn_s_wait_asynccnt(n)
#else
#define WAIT_ASYNC(n) asm volatile("s_wait_asynccnt %0" ::"n"(n))
#endif

// ---------------------------------------------------------------------------
// Row L2-normalization: one wave32 per row of D=256 (8 elems / lane).
// ---------------------------------------------------------------------------
__global__ void __launch_bounds__(256)
vq_norm_rows(const float* __restrict__ src,
             float* __restrict__ dst_f32,       // may be nullptr
             _Float16* __restrict__ dst_f16) {
  const int lane = threadIdx.x & 31;
  const int wave = threadIdx.x >> 5;
  const int row  = blockIdx.x * 8 + wave;

  const float* p = src + (size_t)row * D + lane * 8;
  float4 a = *(const float4*)(p);
  float4 b = *(const float4*)(p + 4);
  float s = a.x*a.x + a.y*a.y + a.z*a.z + a.w*a.w
          + b.x*b.x + b.y*b.y + b.z*b.z + b.w*b.w;
#pragma unroll
  for (int m = 16; m >= 1; m >>= 1) s += __shfl_xor(s, m, 32);
  const float r = rsqrtf(s + VQ_EPS);

  a.x *= r; a.y *= r; a.z *= r; a.w *= r;
  b.x *= r; b.y *= r; b.z *= r; b.w *= r;

  if (dst_f32) {
    float* q = dst_f32 + (size_t)row * D + lane * 8;
    *(float4*)(q)     = a;
    *(float4*)(q + 4) = b;
  }
  v8h o;
  o[0] = (_Float16)a.x; o[1] = (_Float16)a.y; o[2] = (_Float16)a.z; o[3] = (_Float16)a.w;
  o[4] = (_Float16)b.x; o[5] = (_Float16)b.y; o[6] = (_Float16)b.z; o[7] = (_Float16)b.w;
  *(v8h*)(dst_f16 + (size_t)row * D + lane * 8) = o;
}

// ---------------------------------------------------------------------------
// Fused GEMM + row-argmax on v_wmma_f32_16x16x32_f16.
// Block = 8 waves, each wave M=32 rows (two register-resident A sets, 128
// VGPRs); all waves share one 16-code B tile staged in LDS by the async
// engine (global_load_async_to_lds_b128, ASYNCcnt), double-buffered:
//   wait asynccnt<=2 ; barrier ; 16 WMMA from ds_load ; barrier ; issue t+2
// Partial argmax merged across K-splits with global_atomic_max_u64 on a
// monotonic packed key.
//
// VGPR layouts per ISA 7.12.2 (wave32):
//  A (16x32 f16): lane L holds row M=L&15; half h=L>>4 gives K = h*8 + {0..7}
//                 in elems 0..7 and K = h*8 + 16 + {0..7} in elems 8..15.
//  B (32x16 f16): lane L holds column N=L&15; half h gives K = h*8 + {0..7}
//                 in elems 0..7 and K = h*8 + 16 + {0..7} in elems 8..15 (same order as A).
//  C/D (16x16 f32): VGPR v holds row M = v + 8*h, column N = L&15.
// ---------------------------------------------------------------------------
__global__ void __launch_bounds__(256)
vq_argmax(const _Float16* __restrict__ zn,
          const _Float16* __restrict__ en,
          unsigned long long* __restrict__ keys) {
  const int tid  = threadIdx.x;
  const int lane = tid & 31;
  const int wave = tid >> 5;
  const int r    = lane & 15;
  const int hi   = lane >> 4;

  const int rb    = blockIdx.x / KSPLIT;         // row-block (256 rows)
  const int ks    = blockIdx.x % KSPLIT;         // K-split slice
  const int m0    = rb * 256 + wave * 32;        // this wave's 32 rows
  const int cbeg  = ks * (KCODES / KSPLIT);
  const int tiles = (KCODES / KSPLIT) / 16;      // 128 code tiles per block

  // Two A sets: rows [m0, m0+16) and [m0+16, m0+32), full D=256 each.
  FragU A0[8], A1[8];
  {
    const _Float16* ap0 = zn + (size_t)(m0 + r) * D + hi * 8;
    const _Float16* ap1 = ap0 + 16 * D;
#pragma unroll
    for (int kb = 0; kb < 8; ++kb) {
      A0[kb].h[0] = *(const v8h*)(ap0 + kb * 32);
      A0[kb].h[1] = *(const v8h*)(ap0 + kb * 32 + 16);
      A1[kb].h[0] = *(const v8h*)(ap1 + kb * 32);
      A1[kb].h[1] = *(const v8h*)(ap1 + kb * 32 + 16);
    }
  }

  float best0[8], best1[8];
  int   bidx0[8], bidx1[8];
#pragma unroll
  for (int v = 0; v < 8; ++v) {
    best0[v] = -2.0f; bidx0[v] = 0;   // any real dot of unit vectors > -2
    best1[v] = -2.0f; bidx1[v] = 0;
  }

#if USE_ASYNC_LDS
  __shared__ _Float16 lds_tile[2][16 * LROW];    // 2 x 16 padded code rows

  // Cooperative async stage of one 8KB tile (16 codes x 512B, contiguous in
  // global): 512 x 16B chunks, 2 per thread, repacked with LROW padding.
  auto issue_tile = [&](int t) {
    const _Float16* g = en + (size_t)(cbeg + t * 16) * D;
    _Float16* l = &lds_tile[t & 1][0];
#pragma unroll
    for (int j = 0; j < 2; ++j) {
      const int q = tid + j * 256;               // chunk id
      __builtin_amdgcn_global_load_async_to_lds_b128(
          (gptr_v4i)(g + q * 8),
          (lptr_v4i)(l + (q >> 5) * LROW + (q & 31) * 8),
          0, 0);
    }
  };

  issue_tile(0);
  issue_tile(1);

  for (int t = 0; t < tiles; ++t) {
    if (t < tiles - 1) { WAIT_ASYNC(2); } else { WAIT_ASYNC(0); }
    __syncthreads();                             // tile t visible to all waves

    const _Float16* lb = &lds_tile[t & 1][r * LROW + hi * 8];
    v8f acc0 = {}, acc1 = {};
#pragma unroll
    for (int kb = 0; kb < 8; ++kb) {
      FragU B;
      B.h[0] = *(const v8h*)(lb + kb * 32);      // K = h*8 + {0..7}, h*8 + 16 + {0..7}
      B.h[1] = *(const v8h*)(lb + kb * 32 + 16);
      acc0 = __builtin_amdgcn_wmma_f32_16x16x32_f16(
          false, A0[kb].v, false, B.v, (short)0, acc0, false, false);
      acc1 = __builtin_amdgcn_wmma_f32_16x16x32_f16(
          false, A1[kb].v, false, B.v, (short)0, acc1, false, false);
    }

    const int myc = cbeg + t * 16 + r;           // this lane's code column
#pragma unroll
    for (int v = 0; v < 8; ++v) {
      if (acc0[v] > best0[v]) { best0[v] = acc0[v]; bidx0[v] = myc; }  // strict >
      if (acc1[v] > best1[v]) { best1[v] = acc1[v]; bidx1[v] = myc; }  // keeps 1st idx
    }

    __syncthreads();                             // all waves done reading buf
    if (t + 2 < tiles) issue_tile(t + 2);        // overwrite freed buffer
  }
#else
  // Fallback: direct global streaming with a rotating register buffer.
  const _Float16* tp = en + (size_t)(cbeg + r) * D + hi * 8;
  FragU Bb[4];
#pragma unroll
  for (int i = 0; i < 3; ++i) {
    Bb[i].h[0] = *(const v8h*)(tp + i * 32);
    Bb[i].h[1] = *(const v8h*)(tp + i * 32 + 16);
  }
  for (int t = 0; t < tiles; ++t) {
    __builtin_prefetch(tp + 32 * D, 0, 1);
    v8f acc0 = {}, acc1 = {};
#pragma unroll
    for (int kb = 0; kb < 8; ++kb) {
      const int nk = kb + 3;
      const _Float16* p = tp + (nk >> 3) * (16 * D) + (nk & 7) * 32;
      Bb[(kb + 3) & 3].h[0] = *(const v8h*)(p);
      Bb[(kb + 3) & 3].h[1] = *(const v8h*)(p + 16);
      acc0 = __builtin_amdgcn_wmma_f32_16x16x32_f16(
          false, A0[kb].v, false, Bb[kb & 3].v, (short)0, acc0, false, false);
      acc1 = __builtin_amdgcn_wmma_f32_16x16x32_f16(
          false, A1[kb].v, false, Bb[kb & 3].v, (short)0, acc1, false, false);
    }
    const int myc = cbeg + t * 16 + r;
#pragma unroll
    for (int v = 0; v < 8; ++v) {
      if (acc0[v] > best0[v]) { best0[v] = acc0[v]; bidx0[v] = myc; }
      if (acc1[v] > best1[v]) { best1[v] = acc1[v]; bidx1[v] = myc; }
    }
    tp += 16 * D;
  }
#endif

  // Cross-lane reduce within 16-lane halves, then merge K-splits atomically.
#pragma unroll
  for (int v = 0; v < 8; ++v) {
#pragma unroll
    for (int off = 8; off >= 1; off >>= 1) {
      float ov0 = __shfl_xor(best0[v], off, 32);
      int   oi0 = __shfl_xor(bidx0[v], off, 32);
      if (ov0 > best0[v] || (ov0 == best0[v] && oi0 < bidx0[v])) { best0[v] = ov0; bidx0[v] = oi0; }
      float ov1 = __shfl_xor(best1[v], off, 32);
      int   oi1 = __shfl_xor(bidx1[v], off, 32);
      if (ov1 > best1[v] || (ov1 == best1[v] && oi1 < bidx1[v])) { best1[v] = ov1; bidx1[v] = oi1; }
    }
    if (r == v) {
      // dot+4 in [3,5] -> positive f32, bit pattern order-preserving.
      // Low word = ~code so ties pick the SMALLEST code (argmin-first semantics).
      unsigned long long k0 =
          ((unsigned long long)__float_as_uint(best0[v] + 4.0f) << 32) |
          (0xFFFFFFFFu - (unsigned)bidx0[v]);
      unsigned long long k1 =
          ((unsigned long long)__float_as_uint(best1[v] + 4.0f) << 32) |
          (0xFFFFFFFFu - (unsigned)bidx1[v]);
      atomicMax(&keys[m0 + v + 8 * hi], k0);
      atomicMax(&keys[m0 + 16 + v + 8 * hi], k1);
    }
  }
}

// ---------------------------------------------------------------------------
// Gather z_q, write z_q_st = z + (z_q - z), accumulate (z_q - z_norm)^2.
// One wave per row.
// ---------------------------------------------------------------------------
__global__ void __launch_bounds__(256)
vq_finalize(const float* __restrict__ z,
            const float* __restrict__ embn_f32,
            const unsigned long long* __restrict__ keys,
            float* __restrict__ out_zq,     // [NROWS*D]
            float* __restrict__ out_idx,    // [NROWS] (as float, flat f32 concat)
            float* __restrict__ loss_acc) {
  const int lane = threadIdx.x & 31;
  const int wave = threadIdx.x >> 5;
  const int row  = blockIdx.x * 8 + wave;

  const float* zp = z + (size_t)row * D + lane * 8;
  float zv[8];
#pragma unroll
  for (int i = 0; i < 8; ++i) zv[i] = zp[i];

  float s = 0.f;
#pragma unroll
  for (int i = 0; i < 8; ++i) s += zv[i] * zv[i];
#pragma unroll
  for (int m = 16; m >= 1; m >>= 1) s += __shfl_xor(s, m, 32);
  const float rinv = rsqrtf(s + VQ_EPS);

  const int idx = (int)(0xFFFFFFFFu - (unsigned)(keys[row] & 0xFFFFFFFFull));
  const float* ep = embn_f32 + (size_t)idx * D + lane * 8;
  float* op = out_zq + (size_t)row * D + lane * 8;

  float part = 0.f;
#pragma unroll
  for (int i = 0; i < 8; ++i) {
    const float zq = ep[i];
    op[i] = zv[i] + (zq - zv[i]);          // straight-through forward value
    const float d = zq - zv[i] * rinv;     // z_q - z_norm
    part += d * d;
  }
#pragma unroll
  for (int m = 16; m >= 1; m >>= 1) part += __shfl_xor(part, m, 32);
  if (lane == 0) {
    atomicAdd(loss_acc, part);
    out_idx[row] = (float)idx;
  }
}

__global__ void vq_loss_write(const float* __restrict__ loss_acc,
                              float* __restrict__ out_loss) {
  *out_loss = (1.0f + VQ_BETA) * (*loss_acc) / (float)((size_t)NROWS * D);
}

// ---------------------------------------------------------------------------
extern "C" void kernel_launch(void* const* d_in, const int* in_sizes, int n_in,
                              void* d_out, int out_size, void* d_ws, size_t ws_size,
                              hipStream_t stream) {
  const float* z   = (const float*)d_in[0];   // [16,32,32,256]
  const float* emb = (const float*)d_in[1];   // [8192,256]
  float* out = (float*)d_out;                 // [N*D] z_q_st | [1] loss | [N] idx

  char* ws = (char*)d_ws;
  float*    embn_f32 = (float*)ws;                                   // 8 MB
  _Float16* embn_f16 = (_Float16*)(ws + (size_t)8  * 1024 * 1024);   // 4 MB
  _Float16* zn_f16   = (_Float16*)(ws + (size_t)12 * 1024 * 1024);   // 8 MB
  unsigned long long* keys =
      (unsigned long long*)(ws + (size_t)20 * 1024 * 1024);          // 128 KB
  float*    loss_acc = (float*)(ws + (size_t)20 * 1024 * 1024
                                + NROWS * sizeof(unsigned long long));

  // zero argmax keys + loss accumulator (contiguous)
  (void)hipMemsetAsync(keys, 0,
                       NROWS * sizeof(unsigned long long) + sizeof(float), stream);

  vq_norm_rows<<<KCODES / 8, 256, 0, stream>>>(emb, embn_f32, embn_f16);
  vq_norm_rows<<<NROWS  / 8, 256, 0, stream>>>(z, nullptr, zn_f16);
  vq_argmax  <<<(NROWS / 256) * KSPLIT, 256, 0, stream>>>(zn_f16, embn_f16, keys);
  vq_finalize<<<NROWS  / 8, 256, 0, stream>>>(z, embn_f32, keys,
                                              out,
                                              out + (size_t)NROWS * D + 1,
                                              loss_acc);
  vq_loss_write<<<1, 1, 0, stream>>>(loss_acc, out + (size_t)NROWS * D);
}
